// ModConv2d_7791070675251
// MI455X (gfx1250) — compile-verified, matrix operand order checked by reading, not run on hardware
//
#include <hip/hip_runtime.h>

typedef __attribute__((ext_vector_type(16))) __bf16 v16bf;
typedef __attribute__((ext_vector_type(8)))  float  v8f;

#define B_     8
#define C_     512
#define HW_    64
#define AFF_SCALE 0.04419417382415922f      // 1/sqrt(512)
#define CONV_SCALE 0.014731391274719739f    // 1/sqrt(512*9)

#define WT_TILE (9 * 64 * 32)               // 18432 ushorts / buffer
#define IN_TILE (3 * 66 * 32)               //  6336 ushorts / buffer

union FragBF { uint4 u[2]; v16bf v; };

static __device__ __forceinline__ unsigned short f2bf(float f) {
    union { float f; unsigned u; } v; v.f = f;
    unsigned r = v.u + 0x7FFFu + ((v.u >> 16) & 1u);   // round-to-nearest-even
    return (unsigned short)(r >> 16);
}

// ---------------- 1) affine: s[b,i] = style @ wA^T * scale + bA ----------------
__global__ __launch_bounds__(256)
void affine_kernel(const float* __restrict__ style, const float* __restrict__ wA,
                   const float* __restrict__ bA, float* __restrict__ s) {
    int idx = blockIdx.x * 256 + threadIdx.x;           // 4096 = 8*512
    if (idx >= B_ * C_) return;
    int b = idx >> 9, i = idx & 511;
    float acc = 0.f;
    for (int j = 0; j < C_; ++j) acc += style[b * C_ + j] * wA[i * C_ + j];
    s[idx] = acc * AFF_SCALE + bA[i];
}

// ------- 2) modulated input, transposed+bf16: xT[b][y][x][c] = bf16(x * s) -------
__global__ __launch_bounds__(256)
void modin_kernel(const float* __restrict__ in, const float* __restrict__ s,
                  unsigned short* __restrict__ xT) {
    int idx = blockIdx.x * 256 + threadIdx.x;           // 16,777,216 elements
    int c = idx & 511, x = (idx >> 9) & 63, y = (idx >> 15) & 63, b = idx >> 21;
    float v = in[((b * C_ + c) * HW_ + y) * HW_ + x] * s[b * C_ + c];
    xT[idx] = f2bf(v);
}

// ---------------- 3) weights re-laid bf16: wt[tap][oc][ic] ----------------
__global__ __launch_bounds__(256)
void wtr_kernel(const float* __restrict__ wc, unsigned short* __restrict__ wt) {
    int idx = blockIdx.x * 256 + threadIdx.x;           // 2,359,296 = 9*512*512
    int i = idx & 511, o = (idx >> 9) & 511, t = idx >> 18;
    wt[idx] = f2bf(wc[(o * C_ + i) * 9 + t]);
}

// ---------------- 4) dscale[b,o] = rsqrt(cs^2 * Σ_i s² Σ_t w² + eps) * cs ----------------
__global__ __launch_bounds__(256)
void demod_kernel(const float* __restrict__ wc, const float* __restrict__ s,
                  float* __restrict__ dscale) {
    __shared__ float red[256];
    int o = blockIdx.x & 511, b = blockIdx.x >> 9;
    float acc = 0.f;
    for (int i = threadIdx.x; i < C_; i += 256) {
        const float* wp = wc + (o * C_ + i) * 9;
        float w2 = 0.f;
        #pragma unroll
        for (int t = 0; t < 9; ++t) w2 += wp[t] * wp[t];
        float sv = s[b * C_ + i];
        acc += sv * sv * w2;
    }
    red[threadIdx.x] = acc;
    __syncthreads();
    for (int st = 128; st > 0; st >>= 1) {
        if (threadIdx.x < st) red[threadIdx.x] += red[threadIdx.x + st];
        __syncthreads();
    }
    if (threadIdx.x == 0)
        dscale[blockIdx.x] = rsqrtf(CONV_SCALE * CONV_SCALE * red[0] + 1e-8f) * CONV_SCALE;
}

// ---------------- 5) implicit-GEMM 3x3 conv via v_wmma_f32_16x16x32_bf16 ----------------
// grid (y=64, ocTile=8, b=8), block 256 = 8 wave32.
// Block tile: 64 oc x 64 px (one output row). Wave w: M-sub m = w>>1 (16 oc),
// two adjacent 16-px N-subtiles. K-loop: 16 ic-chunks x 9 taps.
// Double-buffered LDS: global loads for chunk k+1 issue before the WMMA loop of
// chunk k, drain to the alternate buffer after, one barrier per chunk.
__global__ __launch_bounds__(256)
void conv_kernel(const unsigned short* __restrict__ xT,   // [8][64][64][512] bf16
                 const unsigned short* __restrict__ wt,   // [9][512][512]   bf16
                 const float* __restrict__ dscale,        // [8][512]
                 float* __restrict__ out) {               // [8][512][64][64]
    __shared__ __align__(16) unsigned short sWt[2][WT_TILE];   // 2 x 36,864 B
    __shared__ __align__(16) unsigned short sIn[2][IN_TILE];   // 2 x 12,672 B

    const int y    = blockIdx.x;
    const int oct  = blockIdx.y;
    const int b    = blockIdx.z;
    const int tid  = threadIdx.x;
    const int lane = tid & 31;
    const int wv   = tid >> 5;
    const int laneM = lane & 15;
    const int hi    = lane >> 4;          // 0: K 0-7/16-23, 1: K 8-15/24-31
    const int m     = wv >> 1;            // M-subtile 0..3
    const int n0    = (wv & 1) * 2;       // N-subtiles n0, n0+1
    const int n1    = n0 + 1;

    v8f acc0 = {};
    v8f acc1 = {};

    uint4 wreg[9];                        // staged weights (2304 uint4 / block)
    uint4 ireg[4];                        // staged input   (792 uint4 / block)

    // ---- stage chunk 0: global -> regs ----
    #pragma unroll
    for (int k = 0; k < 9; ++k) {
        int idx = tid + k * 256;
        int r = idx >> 2, part = idx & 3;            // r = tap*64 + o
        int tap = r >> 6, o = r & 63;
        wreg[k] = *(const uint4*)(wt + ((tap * C_ + oct * 64 + o) * C_ + part * 8));
    }
    #pragma unroll
    for (int k = 0; k < 4; ++k) {
        int idx = tid + k * 256;
        ireg[k] = make_uint4(0u, 0u, 0u, 0u);
        if (idx < 792) {
            int part = idx & 3, pos = idx >> 2;
            int col = pos % 66, ry = pos / 66;
            int x = col - 1, yy = y + ry - 1;
            if ((unsigned)x < 64u && (unsigned)yy < 64u)
                ireg[k] = *(const uint4*)(xT + (((b * HW_ + yy) * HW_ + x) * C_ + part * 8));
        }
    }
    // ---- regs -> LDS buffer 0 ----
    #pragma unroll
    for (int k = 0; k < 9; ++k) {
        int idx = tid + k * 256;
        int r = idx >> 2, part = idx & 3;
        *(uint4*)(&sWt[0][r * 32 + part * 8]) = wreg[k];
    }
    #pragma unroll
    for (int k = 0; k < 4; ++k) {
        int idx = tid + k * 256;
        if (idx < 792) {
            int part = idx & 3, pos = idx >> 2;
            int col = pos % 66, ry = pos / 66;
            *(uint4*)(&sIn[0][(ry * 66 + col) * 32 + part * 8]) = ireg[k];
        }
    }
    __syncthreads();

    for (int icc = 0; icc < 16; ++icc) {
        const int cur = icc & 1;

        // ---- issue global loads for chunk icc+1 (overlaps with WMMA below) ----
        if (icc < 15) {
            const int icn = (icc + 1) * 32;
            #pragma unroll
            for (int k = 0; k < 9; ++k) {
                int idx = tid + k * 256;
                int r = idx >> 2, part = idx & 3;
                int tap = r >> 6, o = r & 63;
                wreg[k] = *(const uint4*)(wt + ((tap * C_ + oct * 64 + o) * C_ + icn + part * 8));
            }
            #pragma unroll
            for (int k = 0; k < 4; ++k) {
                int idx = tid + k * 256;
                ireg[k] = make_uint4(0u, 0u, 0u, 0u);
                if (idx < 792) {
                    int part = idx & 3, pos = idx >> 2;
                    int col = pos % 66, ry = pos / 66;
                    int x = col - 1, yy = y + ry - 1;
                    if ((unsigned)x < 64u && (unsigned)yy < 64u)
                        ireg[k] = *(const uint4*)(xT + (((b * HW_ + yy) * HW_ + x) * C_ + icn + part * 8));
                }
            }
        }

        // ---- compute chunk icc from LDS buffer `cur` ----
        const unsigned short* pw = sWt[cur];
        const unsigned short* pi = sIn[cur];
        #pragma unroll
        for (int tap = 0; tap < 9; ++tap) {
            const int ky = tap / 3, kx = tap % 3;
            // A fragment: 16 oc x 32 ic (two b128 LDS loads per lane)
            FragBF a;
            const int abase = (tap * 64 + m * 16 + laneM) * 32 + hi * 8;
            a.u[0] = *(const uint4*)(pw + abase);
            a.u[1] = *(const uint4*)(pw + abase + 16);
            // B fragments: 32 ic x 16 px, column = x + kx (LDS col 0 == x-1)
            FragBF fb0, fb1;
            const int b0base = ((ky * 66) + (n0 * 16 + laneM + kx)) * 32 + hi * 8;
            fb0.u[0] = *(const uint4*)(pi + b0base);
            fb0.u[1] = *(const uint4*)(pi + b0base + 16);
            const int b1base = ((ky * 66) + (n1 * 16 + laneM + kx)) * 32 + hi * 8;
            fb1.u[0] = *(const uint4*)(pi + b1base);
            fb1.u[1] = *(const uint4*)(pi + b1base + 16);

            acc0 = __builtin_amdgcn_wmma_f32_16x16x32_bf16(
                       false, a.v, false, fb0.v, (short)0, acc0, false, false);
            acc1 = __builtin_amdgcn_wmma_f32_16x16x32_bf16(
                       false, a.v, false, fb1.v, (short)0, acc1, false, false);
        }

        // ---- drain staged regs into the other buffer ----
        if (icc < 15) {
            const int nxt = cur ^ 1;
            #pragma unroll
            for (int k = 0; k < 9; ++k) {
                int idx = tid + k * 256;
                int r = idx >> 2, part = idx & 3;
                *(uint4*)(&sWt[nxt][r * 32 + part * 8]) = wreg[k];
            }
            #pragma unroll
            for (int k = 0; k < 4; ++k) {
                int idx = tid + k * 256;
                if (idx < 792) {
                    int part = idx & 3, pos = idx >> 2;
                    int col = pos % 66, ry = pos / 66;
                    *(uint4*)(&sIn[nxt][(ry * 66 + col) * 32 + part * 8]) = ireg[k];
                }
            }
        }
        __syncthreads();
    }

    // D layout: vgpr r -> M row (hi half of lanes: M+8), lane -> N col
    const int x0 = n0 * 16 + laneM;
    const int x1 = n1 * 16 + laneM;
    #pragma unroll
    for (int r = 0; r < 8; ++r) {
        int oc = oct * 64 + m * 16 + hi * 8 + r;
        float sc = dscale[b * C_ + oc];
        float* orow = out + ((b * C_ + oc) * HW_ + y) * HW_;
        orow[x0] = acc0[r] * sc;
        orow[x1] = acc1[r] * sc;
    }
}

// ---------------------------------------------------------------------------
extern "C" void kernel_launch(void* const* d_in, const int* in_sizes, int n_in,
                              void* d_out, int out_size, void* d_ws, size_t ws_size,
                              hipStream_t stream) {
    const float* input    = (const float*)d_in[0];   // [8,512,64,64]
    const float* style    = (const float*)d_in[1];   // [8,512]
    const float* w_affine = (const float*)d_in[2];   // [512,512]
    const float* b_affine = (const float*)d_in[3];   // [1,512]
    const float* w_conv   = (const float*)d_in[4];   // [512,512,3,3]
    float* outp = (float*)d_out;

    char* ws = (char*)d_ws;
    unsigned short* xT = (unsigned short*)(ws);                         // 33,554,432 B
    unsigned short* wt = (unsigned short*)(ws + 33554432);              //  4,718,592 B
    float* s      = (float*)(ws + 33554432 + 4718592);                  //     16,384 B
    float* dscale = (float*)(ws + 33554432 + 4718592 + 16384);          //     16,384 B

    affine_kernel<<<16, 256, 0, stream>>>(style, w_affine, b_affine, s);
    modin_kernel<<<65536, 256, 0, stream>>>(input, s, xT);
    wtr_kernel<<<9216, 256, 0, stream>>>(w_conv, wt);
    demod_kernel<<<4096, 256, 0, stream>>>(w_conv, s, dscale);

    dim3 grid(HW_, 8, B_);   // y, oc-tile, batch
    conv_kernel<<<grid, 256, 0, stream>>>(xT, wt, dscale, outp);
}
